// FlyGMNetwork_25340307046424
// MI455X (gfx1250) — matrix operand order repaired and checked by reading, not compiled, not run on hardware
//
#include <hip/hip_runtime.h>
#include <hip/hip_bf16.h>
#include <math.h>

typedef __attribute__((ext_vector_type(16))) __bf16 v16bf;
typedef __attribute__((ext_vector_type(8)))  float  v8f;
typedef __attribute__((ext_vector_type(16))) unsigned short v16us;
typedef __attribute__((ext_vector_type(4)))  float  v4f;
typedef __attribute__((ext_vector_type(4)))  int    v4i;

#define NDIM 8192
#define CDIM 32
#define DDIM 16
#define OBSD 75
#define AFFD 512
#define EFF0D 7680
#define BN 4
#define NCOL 128          // BN * CDIM
#define KCH 32            // K per WMMA chunk
#define NKC 256           // total chunks (8192/32)
#define KSPLIT 4
#define KC_PER (NKC / KSPLIT)   // 64
#define MBLK 64                 // rows per block (4 waves x 16)

// workspace layout (bytes)
#define WS_XENC   0                 // 128 f32
#define WS_HID    1024              // 1024 f32 (2 heads * 4 * 128)
#define WS_BFRAG  65536             // 2 MB of u16 (bf16 B fragments)
#define WS_MPART  (4u << 20)        // KSPLIT * 4 MB f32 partials

#if defined(__has_builtin)
#  if __has_builtin(__builtin_amdgcn_global_load_async_to_lds_b128)
#    define HAVE_ASYNC_COPY 1
#  endif
#  if __has_builtin(__builtin_amdgcn_s_wait_asynccnt)
#    define HAVE_WAIT_ASYNC 1
#  endif
#endif

// Builtin param types (probe-verified): arg0 = v4i addrspace(1)*, arg1 = v4i addrspace(3)*.
typedef __attribute__((address_space(1))) v4i* g_v4i_p;
typedef __attribute__((address_space(3))) v4i* l_v4i_p;

__device__ __forceinline__ void wait_async0() {
#if defined(HAVE_WAIT_ASYNC)
  __builtin_amdgcn_s_wait_asynccnt(0);
#elif defined(HAVE_ASYNC_COPY)
  asm volatile("s_wait_asynccnt 0x0" ::: "memory");
#endif
}

// Stage one 8 KB B k-chunk into LDS: 128 threads x 4 x 16B.
// Uses CDNA5 async memory->LDS copies (ASYNCcnt) when available.
__device__ __forceinline__ void stage_chunk(const unsigned short* __restrict__ gsrc,
                                            unsigned short* sdst, int tid) {
  v4i* g4 = (v4i*)gsrc;   // C-style cast drops const
  v4i* s4 = (v4i*)sdst;
#if defined(HAVE_ASYNC_COPY)
#pragma unroll
  for (int i = 0; i < 4; ++i) {
    __builtin_amdgcn_global_load_async_to_lds_b128(
        (g_v4i_p)(g4 + tid + 128 * i),
        (l_v4i_p)(s4 + tid + 128 * i), 0, 0);
  }
#else
#pragma unroll
  for (int i = 0; i < 4; ++i) s4[tid + 128 * i] = g4[tid + 128 * i];
#endif
}

// fp32 -> bf16, round-to-nearest-even (for the pack kernel; scattered stores)
__device__ __forceinline__ unsigned short f2bf(float f) {
  unsigned u = __float_as_uint(f);
  unsigned r = 0x7fffu + ((u >> 16) & 1u);
  return (unsigned short)((u + r) >> 16);
}

// fp32x16 -> bf16x16 via native converts (lowers to v_cvt_pk_bf16_f32, RNE)
__device__ __forceinline__ v16bf cvt_frag(v4f a0, v4f a1, v4f a2, v4f a3) {
  v16bf v;
  v[0]  = (__bf16)a0[0]; v[1]  = (__bf16)a0[1]; v[2]  = (__bf16)a0[2]; v[3]  = (__bf16)a0[3];
  v[4]  = (__bf16)a1[0]; v[5]  = (__bf16)a1[1]; v[6]  = (__bf16)a1[2]; v[7]  = (__bf16)a1[3];
  v[8]  = (__bf16)a2[0]; v[9]  = (__bf16)a2[1]; v[10] = (__bf16)a2[2]; v[11] = (__bf16)a2[3];
  v[12] = (__bf16)a3[0]; v[13] = (__bf16)a3[1]; v[14] = (__bf16)a3[2]; v[15] = (__bf16)a3[3];
  return v;
}

// ---------------- Kernel 1a: LayerNorm + encoder (tiny) ----------------
__global__ void k_enc(const float* __restrict__ obs, const float* __restrict__ ln_g,
                      const float* __restrict__ ln_b, const float* __restrict__ enc_W,
                      const float* __restrict__ enc_b, float* __restrict__ xenc) {
  int t = threadIdx.x;
  if (t >= BN * 32) return;
  int b = t >> 5, e = t & 31;
  const float* o = obs + b * OBSD;
  float mu = 0.f;
  for (int j = 0; j < OBSD; ++j) mu += o[j];
  mu *= (1.0f / OBSD);
  float var = 0.f;
  for (int j = 0; j < OBSD; ++j) { float d = o[j] - mu; var += d * d; }
  var *= (1.0f / OBSD);
  float inv = rsqrtf(var + 1e-5f);
  float acc = enc_b[e];
  for (int j = 0; j < OBSD; ++j) {
    float xn = (o[j] - mu) * inv * ln_g[j] + ln_b[j];
    acc += xn * enc_W[e * OBSD + j];
  }
  xenc[t] = fmaxf(acc, 0.f);
}

// -------- Kernel 1b: gate first 512 rows, pack H2 into bf16 B-fragments --------
// B-fragment layout (16-bit B 32x16, contiguous k order used by this code): for tile (kc, j):
//   lane n (0-15) holds col n, K = kb+0..15 (elements 0..15)
//   lane n+16     holds col n, K = kb+16..31
// Stored flat: Bfrag[((kc*8 + j)*32 + lane)*16 + e]  -> 32 contiguous bytes/lane.
__global__ __launch_bounds__(256) void k_gate_pack(const float* __restrict__ H,
    const float* __restrict__ gate_W, const float* __restrict__ gate_b,
    const float* __restrict__ xenc, unsigned short* __restrict__ Bfrag) {
  int idx = blockIdx.x * 256 + threadIdx.x;   // (u, n)
  int u = idx >> 7, n = idx & 127;
  int b = n >> 5, c = n & 31;
  const float* hrow = H + ((size_t)b * NDIM + u) * CDIM;
  float val;
  if (u < AFFD) {
    float acc = gate_b[c];
    const float* gw = gate_W + c * 64;
    const float* xe = xenc + b * 32;
#pragma unroll
    for (int k = 0; k < 32; ++k) acc += gw[k] * hrow[k];
#pragma unroll
    for (int k = 0; k < 32; ++k) acc += gw[32 + k] * xe[k];
    val = tanhf(acc);
  } else {
    val = hrow[c];
  }
  int kc = u >> 5, r = u & 31;
  int lane = (r < 16) ? (n & 15) : (16 + (n & 15));
  int e = r & 15;
  int j = n >> 4;
  Bfrag[((size_t)((kc * 8 + j) * 32 + lane)) * 16 + e] = f2bf(val);
}

// ---------------- Kernel 2: main GEMM via v_wmma_f32_16x16x32_bf16 ----------------
// C[8192,128] partials = W[8192,8192](fp32->bf16) @ H2frag[8192,128](bf16)
// block = 128 thr (4 waves), each wave owns 16 rows x 128 cols (8 C-tiles).
// grid = (8192/64, KSPLIT); K split across blockIdx.y -> partial sums in ws.
// Double-buffered LDS staging via async memory->LDS copies; 1 barrier/iter.
// B-fragment reads are depth-2 software-pipelined so ds_load latency for
// tile j+1 hides under WMMA j.
__global__ __launch_bounds__(128) void k_gemm(const float* __restrict__ W,
    const unsigned short* __restrict__ Bfrag, float* __restrict__ Mpart) {
  __shared__ unsigned short sB[2][4096];   // 2 x 8 KB (8 tiles * 32 lanes * 16 bf16)
  const int tid  = threadIdx.x;
  const int lane = tid & 31;
  const int wave = tid >> 5;
  const int l15  = lane & 15;
  const int ksel = lane >> 4;
  const int row0 = blockIdx.x * MBLK + wave * 16;
  const int kc0  = blockIdx.y * KC_PER;
  const float* wrow = W + (size_t)(row0 + l15) * NDIM;

  stage_chunk(Bfrag + (size_t)kc0 * 4096, sB[0], tid);

  v8f acc[8] = {};
  for (int kk = 0; kk < KC_PER; ++kk) {
    const int kc  = kc0 + kk;
    const int cur = kk & 1;
    const int kb  = kc * KCH;
    // A fragment: lanes 0-15 K=0..15, lanes 16-31 K=16..31
    // W is a one-shot 256MB stream -> non-temporal to keep B/H_new hot in L2.
    v4f a0 = __builtin_nontemporal_load((const v4f*)(wrow + kb + 16 * ksel));
    v4f a1 = __builtin_nontemporal_load((const v4f*)(wrow + kb + 16 * ksel + 4));
    v4f a2 = __builtin_nontemporal_load((const v4f*)(wrow + kb + 16 * ksel + 8));
    v4f a3 = __builtin_nontemporal_load((const v4f*)(wrow + kb + 16 * ksel + 12));
    if (kk + 4 < KC_PER) __builtin_prefetch(wrow + kb + 4 * KCH, 0, 0);
    wait_async0();          // my buf[cur] async loads complete
    __syncthreads();        // all waves staged buf[cur]; all done reading buf[cur^1]
    if (kk + 1 < KC_PER)
      stage_chunk(Bfrag + (size_t)(kc + 1) * 4096, sB[cur ^ 1], tid);
    v16bf afrag = cvt_frag(a0, a1, a2, a3);
    // tile j fragment lives at sB[cur] + j*512 + lane*16 (u16 units)
    const v16us* bp = (const v16us*)(&sB[cur][0] + lane * 16);  // stride 32 v16us per tile
    v16bf bcur = __builtin_bit_cast(v16bf, bp[0]);
#pragma unroll
    for (int j = 0; j < 8; ++j) {
      v16bf bnxt = bcur;
      if (j < 7) bnxt = __builtin_bit_cast(v16bf, bp[(j + 1) * 32]);  // preload next tile
      acc[j] = __builtin_amdgcn_wmma_f32_16x16x32_bf16(
          false, afrag, false, bcur, (short)0, acc[j], false, false);
      bcur = bnxt;
    }
  }
  // C/D layout: lane(0-15)->N=lane, VGPR g -> M=g; lanes 16-31 -> M=g+8
  float* mp = Mpart + (size_t)blockIdx.y * ((size_t)NDIM * NCOL);
#pragma unroll
  for (int j = 0; j < 8; ++j) {
#pragma unroll
    for (int g = 0; g < 8; ++g) {
      __builtin_nontemporal_store(acc[j][g],
          &mp[(size_t)(row0 + g + 8 * ksel) * NCOL + 16 * j + l15]);
    }
  }
}

// ---------------- Kernel 2b: K-split reduce + per-node MLP -> H_new ----------------
__global__ __launch_bounds__(256) void k_update(const float* __restrict__ Mpart,
    const float* __restrict__ eta, const float* __restrict__ up1_W,
    const float* __restrict__ up1_b, const float* __restrict__ up2_W,
    const float* __restrict__ up2_b, float* __restrict__ Hnew) {
  __shared__ float sW1[64 * 48];
  __shared__ float sW2[32 * 64];
  __shared__ float sb1[64];
  __shared__ float sb2[32];
  for (int i = threadIdx.x; i < 64 * 48; i += 256) sW1[i] = up1_W[i];
  for (int i = threadIdx.x; i < 32 * 64; i += 256) sW2[i] = up2_W[i];
  if (threadIdx.x < 64) sb1[threadIdx.x] = up1_b[threadIdx.x];
  if (threadIdx.x < 32) sb2[threadIdx.x] = up2_b[threadIdx.x];
  __syncthreads();

  int idx = blockIdx.x * 256 + threadIdx.x;   // (v, b)
  int v = idx >> 2, b = idx & 3;
  float in48[48];
  const float* mp = Mpart + (size_t)v * NCOL + b * 32;
#pragma unroll
  for (int c = 0; c < 32; ++c) {
    float s = __builtin_nontemporal_load(&mp[c]);
#pragma unroll
    for (int p = 1; p < KSPLIT; ++p)
      s += __builtin_nontemporal_load(&mp[(size_t)p * ((size_t)NDIM * NCOL) + c]);
    in48[c] = s;
  }
#pragma unroll
  for (int d = 0; d < DDIM; ++d) in48[32 + d] = eta[v * DDIM + d];

  float h[64];
  for (int o = 0; o < 64; ++o) {
    float a = sb1[o];
    const float* w = sW1 + o * 48;
#pragma unroll
    for (int k = 0; k < 48; ++k) a += w[k] * in48[k];
    h[o] = (a > 0.f) ? a : expm1f(a);   // ELU(alpha=1)
  }
  float* out = Hnew + ((size_t)b * NDIM + v) * CDIM;
  for (int c = 0; c < 32; ++c) {
    float a = sb2[c];
    const float* w = sW2 + c * 64;
#pragma unroll
    for (int k = 0; k < 64; ++k) a += w[k] * h[k];
    out[c] = tanhf(a);
  }
}

// ---------------- Kernel 3: head hidden layers (memory-bound GEMV x 1024) ----------------
__global__ __launch_bounds__(256) void k_head1(const float* __restrict__ Hnew,
    const float* __restrict__ act1_W, const float* __restrict__ act1_b,
    const float* __restrict__ cr1_W, const float* __restrict__ cr1_b,
    float* __restrict__ hid) {
  __shared__ float red[256];
  int head = blockIdx.x >> 7;       // 0 = actor, 1 = critic
  int h = blockIdx.x & 127;
  const float* Wr = (head == 0 ? act1_W : cr1_W) + (size_t)h * 16384;
  float a0 = 0.f, a1 = 0.f, a2 = 0.f, a3 = 0.f;
  // H_eff[b][k] = Hnew[b*262144 + 245760 + k] (contiguous)
  for (int k = threadIdx.x; k < 16384; k += 256) {
    float wv = __builtin_nontemporal_load(&Wr[k]);   // weights: one-shot stream
    a0 += wv * Hnew[0 * 262144 + 245760 + k];
    a1 += wv * Hnew[1 * 262144 + 245760 + k];
    a2 += wv * Hnew[2 * 262144 + 245760 + k];
    a3 += wv * Hnew[3 * 262144 + 245760 + k];
  }
  float bias = (head == 0 ? act1_b : cr1_b)[h];
  float accs[4] = {a0, a1, a2, a3};
  for (int b = 0; b < 4; ++b) {
    red[threadIdx.x] = accs[b];
    __syncthreads();
    for (int s = 128; s > 0; s >>= 1) {
      if (threadIdx.x < s) red[threadIdx.x] += red[threadIdx.x + s];
      __syncthreads();
    }
    if (threadIdx.x == 0) hid[head * 512 + b * 128 + h] = fmaxf(red[0] + bias, 0.f);
    __syncthreads();
  }
}

// ---------------- Kernel 4: final logits + value ----------------
__global__ void k_head2(const float* __restrict__ hid, const float* __restrict__ act2_W,
    const float* __restrict__ act2_b, const float* __restrict__ cr2_W,
    const float* __restrict__ cr2_b, float* __restrict__ out) {
  int t = threadIdx.x;
  if (t < 32) {
    int b = t >> 3, a = t & 7;
    float acc = act2_b[a];
    for (int k = 0; k < 128; ++k) acc += hid[b * 128 + k] * act2_W[a * 128 + k];
    out[t] = acc;                    // logits[b][a]
  } else if (t < 36) {
    int b = t - 32;
    float acc = cr2_b[0];
    for (int k = 0; k < 128; ++k) acc += hid[512 + b * 128 + k] * cr2_W[k];
    out[32 + b] = acc;               // value[b]
  }
}

extern "C" void kernel_launch(void* const* d_in, const int* in_sizes, int n_in,
                              void* d_out, int out_size, void* d_ws, size_t ws_size,
                              hipStream_t stream) {
  const float* obs    = (const float*)d_in[0];
  const float* H      = (const float*)d_in[1];
  const float* W      = (const float*)d_in[2];
  const float* eta    = (const float*)d_in[3];
  const float* ln_g   = (const float*)d_in[4];
  const float* ln_b   = (const float*)d_in[5];
  const float* enc_W  = (const float*)d_in[6];
  const float* enc_b  = (const float*)d_in[7];
  const float* gate_W = (const float*)d_in[8];
  const float* gate_b = (const float*)d_in[9];
  const float* up1_W  = (const float*)d_in[10];
  const float* up1_b  = (const float*)d_in[11];
  const float* up2_W  = (const float*)d_in[12];
  const float* up2_b  = (const float*)d_in[13];
  const float* act1_W = (const float*)d_in[14];
  const float* act1_b = (const float*)d_in[15];
  const float* act2_W = (const float*)d_in[16];
  const float* act2_b = (const float*)d_in[17];
  const float* cr1_W  = (const float*)d_in[18];
  const float* cr1_b  = (const float*)d_in[19];
  const float* cr2_W  = (const float*)d_in[20];
  const float* cr2_b  = (const float*)d_in[21];
  (void)in_sizes; (void)n_in; (void)out_size; (void)ws_size;

  char* ws = (char*)d_ws;
  float* xenc = (float*)(ws + WS_XENC);
  float* hid  = (float*)(ws + WS_HID);
  unsigned short* Bfrag = (unsigned short*)(ws + WS_BFRAG);
  float* Mpart = (float*)(ws + WS_MPART);

  float* out  = (float*)d_out;       // [0..31]=logits, [32..35]=value, [36..]=H_new
  float* Hnew = out + 36;

  k_enc<<<1, 128, 0, stream>>>(obs, ln_g, ln_b, enc_W, enc_b, xenc);
  k_gate_pack<<<(NDIM * NCOL) / 256, 256, 0, stream>>>(H, gate_W, gate_b, xenc, Bfrag);
  dim3 gg(NDIM / MBLK, KSPLIT);
  k_gemm<<<gg, 128, 0, stream>>>(W, Bfrag, Mpart);
  k_update<<<(NDIM * BN) / 256, 256, 0, stream>>>(Mpart, eta, up1_W, up1_b, up2_W, up2_b, Hnew);
  k_head1<<<256, 256, 0, stream>>>(Hnew, act1_W, act1_b, cr1_W, cr1_b, hid);
  k_head2<<<1, 64, 0, stream>>>(hid, act2_W, act2_b, cr2_W, cr2_b, out);
}
